// DenseNetBackbone_82111184765285
// MI455X (gfx1250) — matrix operand order repaired and checked by reading, not compiled, not run on hardware
//
#include <hip/hip_runtime.h>

// ---------------------------------------------------------------------------
// MI455X (gfx1250) DenseNet backbone.
// Convs = implicit GEMM on V_WMMA_F32_16X16X32_F16 (wave32).
// Activations f16 end-to-end; weights pre-converted to f16 (K padded to 32).
// Block tile 64(M) x 128(N) x 32(K); 8 waves as 4(M) x 2(N); 4 WMMA per wave
// per K-step sharing one A fragment. LDS double-buffered: stage tile t+1 while
// computing tile t (one barrier per K-step). Weight tiles go global->LDS via
// GLOBAL_LOAD_ASYNC_TO_LDS_B128 (ASYNCcnt) when the builtin is available.
// ---------------------------------------------------------------------------

typedef __attribute__((ext_vector_type(16))) _Float16 v16h;
typedef __attribute__((ext_vector_type(8)))  _Float16 v8h;
typedef __attribute__((ext_vector_type(8)))  float    v8f;

// Types for the async-to-LDS builtin (per clang diagnostic: param is a pointer
// to a 16-byte int vector; AS1 = global source, AS3 = LDS destination).
typedef int v4i __attribute__((vector_size(16)));
typedef __attribute__((address_space(1))) v4i* gas_v4i_p;
typedef __attribute__((address_space(3))) v4i* las_v4i_p;

#define CONV_BM 64
#define CONV_BN 128
#define CONV_BK 32
#define LDS_PAD 8

#if __has_builtin(__builtin_amdgcn_global_load_async_to_lds_b128) && \
    __has_builtin(__builtin_amdgcn_s_wait_asynccnt)
#define USE_ASYNC_LDS 1
#else
#define USE_ASYNC_LDS 0
#endif

template <typename TIn, int KH_C, int KW_C>
__global__ __launch_bounds__(256)
void conv2d_wmma(const TIn* __restrict__ xin, const _Float16* __restrict__ wt,
                 const float* __restrict__ bias, _Float16* __restrict__ y,
                 int Bn, int Cin, int H, int W, int CinTot, int cioff,
                 int Cout, int stride, int pad, int OH, int OW, int Kpad,
                 int CoutTot, int coff, int do_relu)
{
    __shared__ alignas(16) _Float16 ldsA[2][CONV_BM][CONV_BK + LDS_PAD];
    __shared__ alignas(16) _Float16 ldsB[2][CONV_BN][CONV_BK + LDS_PAD];

    const int tid  = threadIdx.x;
    const int lane = tid & 31;
    const int wave = tid >> 5;
    const int wm   = wave & 3;    // 0..3 -> 16-row M sub-tile
    const int wn   = wave >> 2;   // 0..1 -> 64-col N sub-tile

    const int KHW  = KH_C * KW_C;
    const int Ktot = Cin * KHW;
    const int OHW  = OH * OW;
    const int Ntot = Bn * OHW;

    const int Mbase = blockIdx.y * CONV_BM;
    const int Nbase = blockIdx.x * CONV_BN;

    // --- A staging thread map: 4 threads per weight row, 8 f16 each ---
    const int arow = tid >> 2;                 // 0..63
    const int akq  = (tid & 3) * 8;            // 0,8,16,24
    const int gms  = (Mbase + arow) < Cout ? (Mbase + arow) : (Cout - 1);
    const _Float16* wrow = wt + (size_t)gms * Kpad + akq;

    // --- B staging thread map: 2 threads per column, 16 f16 each ---
    const int bcol = tid >> 1;                 // 0..127
    const int bkq  = (tid & 1) * 16;           // 0 or 16
    const int gn   = Nbase + bcol;
    const bool nok = gn < Ntot;
    int nb = 0, ihb = 0, iwb = 0;
    if (nok) {
        nb = gn / OHW;
        int pix = gn - nb * OHW;
        int noh = pix / OW;
        int now = pix - noh * OW;
        ihb = noh * stride - pad;
        iwb = now * stride - pad;
    }
    const long long xbase = ((long long)nb * CinTot + cioff) * (long long)(H * W);

    // ---- staging helpers ----
    auto stageA = [&](int bf, int k0) {
#if USE_ASYNC_LDS
        __builtin_amdgcn_global_load_async_to_lds_b128(
            (gas_v4i_p)(wrow + k0),
            (las_v4i_p)&ldsA[bf][arow][akq],
            0, 0);
#else
        *(v8h*)&ldsA[bf][arow][akq] = *(const v8h*)(wrow + k0);
        if (k0 + CONV_BK < Ktot)
            __builtin_prefetch(wrow + k0 + CONV_BK, 0, 1);   // global_prefetch_b8
#endif
    };
    auto stageB = [&](int bf, int k0) {
        int k  = k0 + bkq;
        int ci = k / KHW;
        int rem = k - ci * KHW;
        int kh = rem / KW_C;
        int kw = rem - kh * KW_C;
        union { _Float16 h[16]; v8h q[2]; } hb;
        #pragma unroll
        for (int j = 0; j < 16; ++j) {
            int ih = ihb + kh;
            int iw = iwb + kw;
            bool ok = nok && ((k + j) < Ktot) &&
                      ih >= 0 && ih < H && iw >= 0 && iw < W;
            long long idx = xbase + (long long)ci * (H * W) + (long long)ih * W + iw;
            idx = ok ? idx : 0;
            TIn raw = xin[idx];                       // unconditional, batched
            hb.h[j] = ok ? (_Float16)raw : (_Float16)0;
            if (++kw == KW_C) { kw = 0; if (++kh == KH_C) { kh = 0; ++ci; } }
        }
        *(v8h*)&ldsB[bf][bcol][bkq]     = hb.q[0];
        *(v8h*)&ldsB[bf][bcol][bkq + 8] = hb.q[1];
    };

    v8f acc[4] = {};
    const int ml   = lane & 15;
    const int half = lane >> 4;
    const int nk   = (Ktot + CONV_BK - 1) / CONV_BK;

    // prologue: stage tile 0 into buffer 0
    stageA(0, 0);
    stageB(0, 0);

    int bf = 0;
    for (int t = 0; t < nk; ++t) {
#if USE_ASYNC_LDS
        __builtin_amdgcn_s_wait_asynccnt(0);   // our async A-tile is in LDS
#endif
        __syncthreads();                        // publish buffer bf

        if (t + 1 < nk) {                       // stage next tile into bf^1
            stageA(bf ^ 1, (t + 1) * CONV_BK);
            stageB(bf ^ 1, (t + 1) * CONV_BK);
        }

        // ---- fragments (CDNA5 16-bit WMMA VGPR layout) + 4 WMMA per wave ----
        union { v16h v; v8h h[2]; } af;
        {
            const _Float16* ar = &ldsA[bf][wm * 16 + ml][0];
            af.h[0] = *(const v8h*)(ar + half * 8);        // K 0..7   / 8..15
            af.h[1] = *(const v8h*)(ar + 16 + half * 8);   // K 16..23 / 24..31
        }
        #pragma unroll
        for (int q = 0; q < 4; ++q) {
            union { v16h v; v8h h[2]; } bfr;
            const _Float16* br = &ldsB[bf][wn * 64 + q * 16 + ml][0];
            bfr.h[0] = *(const v8h*)(br + half * 8);       // K 0..7   / 8..15
            bfr.h[1] = *(const v8h*)(br + 16 + half * 8);  // K 16..23 / 24..31
            acc[q] = __builtin_amdgcn_wmma_f32_16x16x32_f16(
                         false, af.v, false, bfr.v, (short)0, acc[q], false, false);
        }
        bf ^= 1;
    }

    // ---- epilogue: bias + relu fused, channel-offset NCHW f16 store ----
    const int mh = (lane >> 4) * 8;
    #pragma unroll
    for (int q = 0; q < 4; ++q) {
        int n = Nbase + wn * 64 + q * 16 + ml;
        if (n < Ntot) {
            int bb  = n / OHW;
            int pix = n - bb * OHW;
            #pragma unroll
            for (int r = 0; r < 8; ++r) {
                int m = Mbase + wm * 16 + mh + r;
                if (m < Cout) {
                    float v = acc[q][r];
                    if (bias) v += bias[m];
                    if (do_relu && v < 0.0f) v = 0.0f;
                    y[((size_t)bb * CoutTot + coff + m) * OHW + pix] = (_Float16)v;
                }
            }
        }
    }
}

// ---------------------------------------------------------------------------
// Weight pre-conversion: f32 [Cout][Ktot] -> f16 [Cout][Kpad], zero-padded.
// ---------------------------------------------------------------------------
__global__ void cvt_w_k(const float* __restrict__ w, _Float16* __restrict__ o,
                        int Cout, int Ktot, int Kpad, int total)
{
    for (int i = blockIdx.x * blockDim.x + threadIdx.x; i < total;
         i += gridDim.x * blockDim.x) {
        int m = i / Kpad, k = i - m * Kpad;
        o[i] = (k < Ktot) ? (_Float16)w[(size_t)m * Ktot + k] : (_Float16)0;
    }
}

// ---------------------------------------------------------------------------
// Pointwise / glue kernels. f16 activations, f32 math, f32 params.
// ---------------------------------------------------------------------------
__global__ void bnrelu_k(const _Float16* __restrict__ x, const float* __restrict__ g,
                         const float* __restrict__ b, const float* __restrict__ m,
                         const float* __restrict__ v, _Float16* __restrict__ y,
                         int C, int HW, int CinTot, int cioff, int total)
{
    for (int i = blockIdx.x * blockDim.x + threadIdx.x; i < total;
         i += gridDim.x * blockDim.x) {
        int p = i % HW; int t = i / HW; int c = t % C; int bb = t / C;
        float xv  = (float)x[((size_t)bb * CinTot + cioff + c) * HW + p];
        float inv = rsqrtf(v[c] + 1e-5f);
        float r   = (xv - m[c]) * (g[c] * inv) + b[c];
        y[i] = (_Float16)(r > 0.0f ? r : 0.0f);
    }
}

__global__ void maxpool3s2_k(const _Float16* __restrict__ x, _Float16* __restrict__ y,
                             int C, int H, int W, int OH, int OW,
                             int CoutTot, int coff, int total)
{
    for (int i = blockIdx.x * blockDim.x + threadIdx.x; i < total;
         i += gridDim.x * blockDim.x) {
        int ow = i % OW; int t = i / OW; int oh = t % OH; t /= OH;
        int c = t % C; int bb = t / C;
        float best = -3.402823e38f;
        for (int dy = 0; dy < 3; ++dy) {
            int ih = oh * 2 - 1 + dy;
            if (ih < 0 || ih >= H) continue;
            for (int dx = 0; dx < 3; ++dx) {
                int iw = ow * 2 - 1 + dx;
                if (iw < 0 || iw >= W) continue;
                float v = (float)x[(((size_t)bb * C + c) * H + ih) * W + iw];
                if (v > best) best = v;
            }
        }
        y[(((size_t)bb * CoutTot + coff + c) * OH + oh) * OW + ow] = (_Float16)best;
    }
}

__global__ void avgpool_k(const _Float16* __restrict__ x, _Float16* __restrict__ y,
                          int C, int H, int W, int k,
                          int CoutTot, int coff, int total)
{
    int OH = H / k, OW = W / k;
    for (int i = blockIdx.x * blockDim.x + threadIdx.x; i < total;
         i += gridDim.x * blockDim.x) {
        int ow = i % OW; int t = i / OW; int oh = t % OH; t /= OH;
        int c = t % C; int bb = t / C;
        float s = 0.0f;
        for (int dy = 0; dy < 8; ++dy) {
            if (dy >= k) break;
            for (int dx = 0; dx < 8; ++dx) {
                if (dx >= k) break;
                s += (float)x[(((size_t)bb * C + c) * H + oh * k + dy) * W + ow * k + dx];
            }
        }
        y[(((size_t)bb * CoutTot + coff + c) * OH + oh) * OW + ow] =
            (_Float16)(s / (float)(k * k));
    }
}

__global__ void sigmoid_k(_Float16* __restrict__ x, int total)
{
    for (int i = blockIdx.x * blockDim.x + threadIdx.x; i < total;
         i += gridDim.x * blockDim.x)
        x[i] = (_Float16)(1.0f / (1.0f + expf(-(float)x[i])));
}

// Deformable-conv bilinear gather -> expanded [B, C, 3H, 3W] (f16).
__global__ void deform_sample_k(const _Float16* __restrict__ x,
                                const _Float16* __restrict__ off,
                                const _Float16* __restrict__ msk,
                                _Float16* __restrict__ y,
                                int C, int H, int W, int total)
{
    int EW = 3 * W, EH = 3 * H;
    int Hp = H + 2, Wp = W + 2;
    for (int i = blockIdx.x * blockDim.x + threadIdx.x; i < total;
         i += gridDim.x * blockDim.x) {
        int ox = i % EW; int t = i / EW; int oy = t % EH; t /= EH;
        int c = t % C; int bb = t / C;
        int w0 = ox / 3, s = ox - w0 * 3;
        int h0 = oy / 3, r = oy - h0 * 3;
        int n = r * 3 + s;
        float py = (float)off[(((size_t)bb * 18 + n) * H + h0) * W + w0] +
                   (float)(r - 1) + (float)(h0 + 1);
        float px = (float)off[(((size_t)bb * 18 + 9 + n) * H + h0) * W + w0] +
                   (float)(s - 1) + (float)(w0 + 1);
        float pyf = floorf(py), pxf = floorf(px);
        int q0y = (int)fminf(fmaxf(pyf, 0.0f), (float)(Hp - 1));
        int q0x = (int)fminf(fmaxf(pxf, 0.0f), (float)(Wp - 1));
        int q1y = (int)fminf(fmaxf(pyf + 1.0f, 0.0f), (float)(Hp - 1));
        int q1x = (int)fminf(fmaxf(pxf + 1.0f, 0.0f), (float)(Wp - 1));
        float pyc = fminf(fmaxf(py, 0.0f), (float)(Hp - 1));
        float pxc = fminf(fmaxf(px, 0.0f), (float)(Wp - 1));
        float g_lt = (1.0f + (q0y - pyc)) * (1.0f + (q0x - pxc));
        float g_rb = (1.0f - (q1y - pyc)) * (1.0f - (q1x - pxc));
        float g_lb = (1.0f + (q0y - pyc)) * (1.0f - (q1x - pxc));
        float g_rt = (1.0f - (q1y - pyc)) * (1.0f + (q0x - pxc));
        const _Float16* xb = x + ((size_t)bb * C + c) * H * W;
        auto samp = [&](int qy, int qx) -> float {
            if (qy >= 1 && qy <= H && qx >= 1 && qx <= W)
                return (float)xb[(qy - 1) * W + (qx - 1)];
            return 0.0f;
        };
        float v = g_lt * samp(q0y, q0x) + g_rb * samp(q1y, q1x) +
                  g_lb * samp(q0y, q1x) + g_rt * samp(q1y, q0x);
        v *= (float)msk[(((size_t)bb * 9 + n) * H + h0) * W + w0];
        y[(((size_t)bb * C + c) * EH + oy) * EW + ox] = (_Float16)v;
    }
}

// Transpose conv (stride 2, pad 1, k=3, transpose_kernel): w f32 [Cin][Cout][3][3].
__global__ void convT_k(const _Float16* __restrict__ x, const float* __restrict__ w,
                        const float* __restrict__ bias, _Float16* __restrict__ y,
                        int Cin, int H, int W, int Cout, int OH, int OW, int total)
{
    for (int i = blockIdx.x * blockDim.x + threadIdx.x; i < total;
         i += gridDim.x * blockDim.x) {
        int ox = i % OW; int t = i / OW; int oy = t % OH; t /= OH;
        int oc = t % Cout; int bb = t / Cout;
        float s = bias ? bias[oc] : 0.0f;
        for (int kh = 0; kh < 3; ++kh) {
            int ty = oy + 1 - kh;
            if (ty < 0 || (ty & 1)) continue;
            int iy = ty >> 1; if (iy >= H) continue;
            for (int kw = 0; kw < 3; ++kw) {
                int tx = ox + 1 - kw;
                if (tx < 0 || (tx & 1)) continue;
                int ix = tx >> 1; if (ix >= W) continue;
                const _Float16* xb = x + (((size_t)bb * Cin) * H + iy) * W + ix;
                const float* wb = w + ((size_t)oc * 9 + kh * 3 + kw);
                for (int ci = 0; ci < Cin; ++ci)
                    s += (float)xb[(size_t)ci * H * W] * wb[(size_t)ci * Cout * 9];
            }
        }
        y[(((size_t)bb * Cout + oc) * OH + oy) * OW + ox] = (_Float16)s;
    }
}

// Align-corners bilinear upsample (TOut = f16 mid-net, float for final output).
template <typename TOut>
__global__ void upsample_k(const _Float16* __restrict__ x, TOut* __restrict__ y,
                           int C, int H, int W, int OH, int OW, int total)
{
    for (int i = blockIdx.x * blockDim.x + threadIdx.x; i < total;
         i += gridDim.x * blockDim.x) {
        int ox = i % OW; int t = i / OW; int oy = t % OH; t /= OH;
        int c = t % C; int bb = t / C;
        float ys = (float)oy * (float)(H - 1) / (float)(OH - 1);
        float xs = (float)ox * (float)(W - 1) / (float)(OW - 1);
        float y0f = floorf(ys), x0f = floorf(xs);
        int y0 = (int)y0f, x0 = (int)x0f;
        int y1 = y0 + 1 < H ? y0 + 1 : H - 1;
        int x1 = x0 + 1 < W ? x0 + 1 : W - 1;
        float wy = ys - y0f, wx = xs - x0f;
        const _Float16* xb = x + ((size_t)bb * C + c) * H * W;
        float v = (float)xb[y0 * W + x0] * (1 - wy) * (1 - wx) +
                  (float)xb[y0 * W + x1] * (1 - wy) * wx +
                  (float)xb[y1 * W + x0] * wy * (1 - wx) +
                  (float)xb[y1 * W + x1] * wy * wx;
        y[(((size_t)bb * C + c) * OH + oy) * OW + ox] = (TOut)v;
    }
}

// ---------------------------------------------------------------------------
// Host orchestration.
// Param flattening assumption: x = d_in[0], d = d_in[1], then params leaves in
// sorted-key pytree order (dicts sorted alphabetically, lists in order).
// ---------------------------------------------------------------------------
static inline int ceil_div(int a, int b) { return (a + b - 1) / b; }

extern "C" void kernel_launch(void* const* d_in, const int* in_sizes, int n_in,
                              void* d_out, int out_size, void* d_ws, size_t ws_size,
                              hipStream_t stream)
{
    (void)in_sizes; (void)n_in; (void)out_size; (void)ws_size;
    const int Bn = 8;
    const float* X = (const float*)d_in[0];
    const float* D = (const float*)d_in[1];
    auto P = [&](int i) { return (const float*)d_in[i]; };

    // bump allocator over d_ws
    char* ap = (char*)d_ws;
    auto alloc_bytes = [&](size_t b) { char* r = ap; ap += ((b + 255) & ~(size_t)255); return r; };
    auto allocH = [&](size_t elems) { return (_Float16*)alloc_bytes(elems * 2); };

    // ---- weight pre-conversion (f32 -> f16, K padded to 32) ----
    auto wprep = [&](int idx, int Cout, int Ktot) -> _Float16* {
        int Kpad = (Ktot + 31) & ~31;
        _Float16* o = allocH((size_t)Cout * Kpad);
        int total = Cout * Kpad;
        cvt_w_k<<<ceil_div(total, 256), 256, 0, stream>>>(P(idx), o, Cout, Ktot, Kpad, total);
        return o;
    };

    _Float16* W_conv0 = wprep(2, 64, 3 * 49);
    _Float16 *W_db1c1[6], *W_db1c2[6], *W_db2c1[12], *W_db2c2[12];
    for (int l = 0; l < 6; ++l) {
        W_db1c1[l] = wprep(18 + l * 10 + 8, 128, 64 + 32 * l);
        W_db1c2[l] = wprep(18 + l * 10 + 9, 32, 128 * 9);
    }
    for (int l = 0; l < 12; ++l) {
        W_db2c1[l] = wprep(78 + l * 10 + 8, 128, 128 + 32 * l);
        W_db2c2[l] = wprep(78 + l * 10 + 9, 32, 128 * 9);
    }
    _Float16* W_tr  = wprep(222, 128, 256);
    _Float16* W_l1a = wprep(205, 256, 512 * 9);
    _Float16* W_l1b = wprep(207, 128, 256 * 9);
    _Float16* W_l1c = wprep(209, 32, 128 * 9);
    _Float16* W_p0  = wprep(215, 32, 1 * 9);
    _Float16* W_p1  = wprep(217, 32, 32 * 9);
    _Float16* W_dp[3], *W_dm[3], *W_dw[3];
    const int dbase[3] = {3, 8, 13};      // d1,d2,d3: mb,mw,pb,pw,w
    const int dcin[3]  = {64, 128, 128};
    const int dcout[3] = {128, 128, 64};
    for (int i = 0; i < 3; ++i) {
        W_dm[i] = wprep(dbase[i] + 1, 9,  dcin[i] * 9);
        W_dp[i] = wprep(dbase[i] + 3, 18, dcin[i] * 9);
        W_dw[i] = wprep(dbase[i] + 4, dcout[i], dcin[i] * 9);
    }

    // ---- conv dispatch ----
    auto conv = [&](const void* x, bool xf32, const _Float16* w, const float* bias,
                    _Float16* y, int Cin, int H, int W, int CinTot, int cioff,
                    int Cout, int KH, int KW, int st, int pad,
                    int CoutTot, int coff, int relu) {
        int OH = (H + 2 * pad - KH) / st + 1;
        int OW = (W + 2 * pad - KW) / st + 1;
        int Ntot = Bn * OH * OW;
        int Kpad = (Cin * KH * KW + 31) & ~31;
        dim3 g(ceil_div(Ntot, CONV_BN), ceil_div(Cout, CONV_BM));
        if (xf32) {
            if (KH == 7)
                conv2d_wmma<float, 7, 7><<<g, 256, 0, stream>>>(
                    (const float*)x, w, bias, y, Bn, Cin, H, W, CinTot, cioff,
                    Cout, st, pad, OH, OW, Kpad, CoutTot, coff, relu);
            else
                conv2d_wmma<float, 3, 3><<<g, 256, 0, stream>>>(
                    (const float*)x, w, bias, y, Bn, Cin, H, W, CinTot, cioff,
                    Cout, st, pad, OH, OW, Kpad, CoutTot, coff, relu);
        } else {
            if (KH == 1)
                conv2d_wmma<_Float16, 1, 1><<<g, 256, 0, stream>>>(
                    (const _Float16*)x, w, bias, y, Bn, Cin, H, W, CinTot, cioff,
                    Cout, st, pad, OH, OW, Kpad, CoutTot, coff, relu);
            else
                conv2d_wmma<_Float16, 3, 3><<<g, 256, 0, stream>>>(
                    (const _Float16*)x, w, bias, y, Bn, Cin, H, W, CinTot, cioff,
                    Cout, st, pad, OH, OW, Kpad, CoutTot, coff, relu);
        }
    };
    auto bnrelu = [&](const _Float16* x, int gi, _Float16* y, int C, int HW,
                      int CinTot, int cioff) {
        int total = Bn * C * HW;   // bn leaves sorted: b,g,m,v
        bnrelu_k<<<ceil_div(total, 256), 256, 0, stream>>>(
            x, P(gi + 1), P(gi + 0), P(gi + 2), P(gi + 3), y, C, HW, CinTot, cioff, total);
    };

    // ---------------- stem ----------------
    _Float16* A0 = allocH((size_t)Bn * 64 * 112 * 112);
    _Float16* A1 = allocH((size_t)Bn * 64 * 112 * 112);
    conv(X, true, W_conv0, nullptr, A0, 3, 224, 224, 3, 0, 64, 7, 7, 2, 3, 64, 0, 0);
    bnrelu(A0, 210, A1, 64, 112 * 112, 64, 0);
    _Float16* cur1 = A0;                       // [8,256,56,56] fits (A0 dead)
    {
        int total = Bn * 64 * 56 * 56;
        maxpool3s2_k<<<ceil_div(total, 256), 256, 0, stream>>>(
            A1, cur1, 64, 112, 112, 56, 56, 256, 0, total);
    }

    _Float16* tbn  = allocH((size_t)Bn * 256 * 56 * 56);
    _Float16* tc1  = allocH((size_t)Bn * 128 * 56 * 56);
    _Float16* tbn2 = allocH((size_t)Bn * 128 * 56 * 56);

    auto dense_block = [&](_Float16* cur, int Ctot, int C0, int nlayers,
                           int baseIdx, _Float16** Wc1, _Float16** Wc2,
                           int H, int W) {
        int C = C0;
        for (int l = 0; l < nlayers; ++l) {
            int bi = baseIdx + l * 10;
            bnrelu(cur, bi + 0, tbn, C, H * W, Ctot, 0);
            conv(tbn, false, Wc1[l], nullptr, tc1, C, H, W, C, 0,
                 128, 1, 1, 1, 0, 128, 0, 0);
            bnrelu(tc1, bi + 4, tbn2, 128, H * W, 128, 0);
            conv(tbn2, false, Wc2[l], nullptr, cur, 128, H, W, 128, 0,
                 32, 3, 3, 1, 1, Ctot, C, 0);
            C += 32;
        }
    };

    // ---------------- dense block 1 (64 -> 256 @56) ----------------
    dense_block(cur1, 256, 64, 6, 18, W_db1c1, W_db1c2, 56, 56);

    // ---------------- transition ----------------
    bnrelu(cur1, 218, tbn, 256, 56 * 56, 256, 0);
    conv(tbn, false, W_tr, nullptr, tc1, 256, 56, 56, 256, 0, 128, 1, 1, 1, 0, 128, 0, 0);
    _Float16* cur2 = A1;                        // [8,512,28,28] fits (A1 dead)
    {
        int total = Bn * 128 * 28 * 28;
        avgpool_k<<<ceil_div(total, 256), 256, 0, stream>>>(
            tc1, cur2, 128, 56, 56, 2, 512, 0, total);
    }

    // ---------------- dense block 2 (128 -> 512 @28) ----------------
    dense_block(cur2, 512, 128, 12, 78, W_db2c1, W_db2c2, 28, 28);

    // ---------------- layer1 ----------------
    _Float16* t_a = tbn;
    _Float16* t_b = tbn2;
    _Float16* cat = allocH((size_t)Bn * 64 * 28 * 28);
    conv(cur2, false, W_l1a, P(204), t_a, 512, 28, 28, 512, 0, 256, 3, 3, 1, 1, 256, 0, 0);
    conv(t_a,  false, W_l1b, P(206), t_b, 256, 28, 28, 256, 0, 128, 3, 3, 1, 1, 128, 0, 0);
    conv(t_b,  false, W_l1c, P(208), cat, 128, 28, 28, 128, 0, 32, 3, 3, 1, 1, 64, 0, 0);

    // ---------------- depth branch ----------------
    _Float16* p0 = allocH((size_t)Bn * 32 * 224 * 224);
    _Float16* p1 = allocH((size_t)Bn * 32 * 224 * 224);
    conv(D,  true,  W_p0, P(214), p0, 1, 224, 224, 1, 0, 32, 3, 3, 1, 1, 32, 0, 0);
    conv(p0, false, W_p1, P(216), p1, 32, 224, 224, 32, 0, 32, 3, 3, 1, 1, 32, 0, 0);
    {
        int total = Bn * 32 * 28 * 28;
        avgpool_k<<<ceil_div(total, 256), 256, 0, stream>>>(
            p1, cat, 32, 224, 224, 8, 64, 32, total);
    }

    // ---------------- deformable convs ----------------
    _Float16* toff  = allocH((size_t)Bn * 18 * 28 * 28);
    _Float16* tmsk  = allocH((size_t)Bn * 9 * 28 * 28);
    _Float16* txoff = allocH((size_t)Bn * 128 * 84 * 84);

    auto deform = [&](const _Float16* xin, int C, int Cout, int di, _Float16* out) {
        conv(xin, false, W_dp[di], P(dbase[di] + 2), toff, C, 28, 28, C, 0,
             18, 3, 3, 1, 1, 18, 0, 0);
        conv(xin, false, W_dm[di], P(dbase[di] + 0), tmsk, C, 28, 28, C, 0,
             9, 3, 3, 1, 1, 9, 0, 0);
        int tm = Bn * 9 * 28 * 28;
        sigmoid_k<<<ceil_div(tm, 256), 256, 0, stream>>>(tmsk, tm);
        int ts = Bn * C * 84 * 84;
        deform_sample_k<<<ceil_div(ts, 256), 256, 0, stream>>>(
            xin, toff, tmsk, txoff, C, 28, 28, ts);
        conv(txoff, false, W_dw[di], nullptr, out, C, 84, 84, C, 0,
             Cout, 3, 3, 3, 0, Cout, 0, 0);
    };

    _Float16* h1 = tc1;
    _Float16* h2 = tbn2;
    _Float16* h3 = tbn;
    deform(cat, 64, 128, 0, h1);
    deform(h1, 128, 128, 1, h2);
    deform(h2, 128, 64, 2, h3);

    // ---------------- decoder ----------------
    _Float16* t4 = A1;   // [8,64,55,55]
    _Float16* t5 = p0;   // [8,128,109,109]
    _Float16* t6 = p1;   // [8,128,112,112]
    _Float16* t7 = A0;   // [8,1,223,223]
    {
        int total = Bn * 64 * 55 * 55;
        convT_k<<<ceil_div(total, 256), 256, 0, stream>>>(
            h3, P(199), P(198), t4, 64, 28, 28, 64, 55, 55, total);
    }
    {
        int total = Bn * 128 * 109 * 109;
        convT_k<<<ceil_div(total, 256), 256, 0, stream>>>(
            t4, P(201), P(200), t5, 64, 55, 55, 128, 109, 109, total);
    }
    {
        int total = Bn * 128 * 112 * 112;
        upsample_k<_Float16><<<ceil_div(total, 256), 256, 0, stream>>>(
            t5, t6, 128, 109, 109, 112, 112, total);
    }
    {
        int total = Bn * 1 * 223 * 223;
        convT_k<<<ceil_div(total, 256), 256, 0, stream>>>(
            t6, P(203), P(202), t7, 128, 112, 112, 1, 223, 223, total);
    }
    {
        int total = Bn * 1 * 224 * 224;
        upsample_k<float><<<ceil_div(total, 256), 256, 0, stream>>>(
            t7, (float*)d_out, 1, 223, 223, 224, 224, total);
    }
}
